// Atom_Atom_embedding_MP_19988777795863
// MI455X (gfx1250) — matrix operand order repaired and checked by reading, not compiled, not run on hardware
//
#include <hip/hip_runtime.h>

#define NPTS 16384
#define DF   16      // feature dim
#define KNBR 16      // neighbors kept (self dropped by index exclusion)
#define DIN  33      // 2*DF+1
#define NEGS 0.2f

typedef __attribute__((ext_vector_type(16))) _Float16 v16h;
typedef __attribute__((ext_vector_type(8)))  float    v8f;
typedef __attribute__((ext_vector_type(2)))  float    v2f;

// ---------------------------------------------------------------------------
// Prep: P[i][8] = { x, y, z, |x|^2,  -2x, -2y, -2z, 1 }  (augmented GEMM trick)
// ---------------------------------------------------------------------------
__global__ void prep_P(const float* __restrict__ x, const float* __restrict__ y,
                       float* __restrict__ P) {
  int i = blockIdx.x * blockDim.x + threadIdx.x;
  if (i >= NPTS) return;
  float x0 = x[3*i], x1 = x[3*i+1], x2 = x[3*i+2];
  float y0 = y[3*i], y1 = y[3*i+1], y2 = y[3*i+2];
  float* p = P + (size_t)i * 8;
  p[0] = x0; p[1] = x1; p[2] = x2; p[3] = x0*x0 + x1*x1 + x2*x2;
  p[4] = -2.f*y0; p[5] = -2.f*y1; p[6] = -2.f*y2; p[7] = 1.0f;
}

// batch boundaries (batches are sorted, values 0..3): bstart[0..4]
__global__ void prep_batch(const int* __restrict__ xb, int* __restrict__ bstart) {
  int i = blockIdx.x * blockDim.x + threadIdx.x;
  if (i >= NPTS) return;
  if (i == 0) { int b0 = xb[0]; for (int b = 0; b <= b0; ++b) bstart[b] = 0; }
  else { int pb = xb[i-1], cb = xb[i]; for (int b = pb+1; b <= cb; ++b) bstart[b] = i; }
  if (i == NPTS-1) { for (int b = xb[i]+1; b <= 4; ++b) bstart[b] = NPTS; }
}

// ---------------------------------------------------------------------------
// Fused KNN: one wave owns a 16-row query tile; each column tile's 16x16 d^2
// block is ONE v_wmma_f32_16x16x4_f32. Column-tile P data (512B) is staged
// into per-wave LDS by a double-buffered global_load_async_to_lds_b128
// pipeline (ASYNCcnt). Row r is owned by lanes (r, r+16), each keeping an
// exact register-sorted top-16 of its column-residue half; batch masking is
// pure VALU via per-row [rowLo,rowHi) (batches are sorted).
// ---------------------------------------------------------------------------
#define KNN_WAVES 4
__global__ __launch_bounds__(128)
void knn_kernel(const float* __restrict__ P, const int* __restrict__ xb,
                const int* __restrict__ bstart,
                int* __restrict__ idxOut, float* __restrict__ d2Out) {
  __shared__ float s_pb[KNN_WAVES][2][128];      // async-staged P column tiles
  __shared__ float s_tile[KNN_WAVES][16 * 20];   // padded d2 tile (b128-aligned rows)
  __shared__ float s_md[KNN_WAVES][16 * 34];     // merge lists (16+1 sentinel) x2
  __shared__ int   s_mi[KNN_WAVES][16 * 34];

  const int w    = threadIdx.x >> 5;
  const int lane = threadIdx.x & 31;
  const int half = lane >> 4;
  const int l15  = lane & 15;
  const int rowBase = (blockIdx.x * KNN_WAVES + w) << 4;
  const int row  = rowBase + l15;
  const int rb   = xb[row];
  const int rowLo = bstart[rb];          // this row's valid column range
  const int rowHi = bstart[rb + 1];

  // A matrix (16x4 f32): lanes 0-15 K={0,1}, lanes 16-31 K={2,3}, M=lane&15
  v2f a;
  { const float* p = P + (size_t)row * 8 + half * 2; a[0] = p[0]; a[1] = p[1]; }

  float bd[16]; int bi[16];
#pragma unroll
  for (int s = 0; s < 16; ++s) { bd[s] = 3.0e38f; bi[s] = row; }

  const int cLo = bstart[xb[rowBase]];           // wave sweep range (union)
  const int cHi = bstart[xb[rowBase + 15] + 1];
  const int t0 = cLo >> 4, t1 = (cHi + 15) >> 4;

  // async stage of one 512B column tile: 32 lanes x 16B, mem -> LDS
  auto stage = [&](int tt, int bufi) {
    const float* g = P + ((size_t)(tt << 4)) * 8 + lane * 4;
    unsigned loff = (unsigned)(uintptr_t)&s_pb[w][bufi][lane * 4];
    asm volatile("global_load_async_to_lds_b128 %0, %1, off"
                 :: "v"(loff), "v"((unsigned long long)(uintptr_t)g)
                 : "memory");
  };

  stage(t0, t0 & 1);
  for (int t = t0; t < t1; ++t) {
    const int buf = t & 1;
    const int colBase = t << 4;
    if (t + 1 < t1) {
      stage(t + 1, buf ^ 1);                          // overlap next copy
      asm volatile("s_wait_asynccnt 0x1" ::: "memory"); // tile t ready (in-order)
    } else {
      asm volatile("s_wait_asynccnt 0x0" ::: "memory");
    }

    // B matrix (4x16 f32): lanes 0-15 K={0,1}, lanes 16-31 K={2,3}, N=lane&15
    v2f b = *(const v2f*)&s_pb[w][buf][l15 * 8 + 4 + half * 2];
    const float cv = s_pb[w][buf][l15 * 8 + 3];       // |y_col|^2
    v8f c;
#pragma unroll
    for (int r = 0; r < 8; ++r) c[r] = cv;

    v8f dm = __builtin_amdgcn_wmma_f32_16x16x4_f32(
        false, a, false, b, (short)0, c, false, false);

    // D layout: VGPR r -> row r (lanes 0-15) / row r+8 (lanes 16-31), col=lane&15
#pragma unroll
    for (int r = 0; r < 8; ++r)
      s_tile[w][(r + 8 * half) * 20 + l15] = dm[r];
    asm volatile("s_wait_dscnt 0x0" ::: "memory");

    if (t + 2 < t1)
      __builtin_prefetch(P + ((size_t)((t + 2) << 4)) * 8, 0, 1);

    // owner scan: lane (half,l15) consumes columns colBase+8*half .. +7
    const float4 qa = *(const float4*)&s_tile[w][l15 * 20 + half * 8];
    const float4 qb = *(const float4*)&s_tile[w][l15 * 20 + half * 8 + 4];
    const float dv8[8] = {qa.x, qa.y, qa.z, qa.w, qb.x, qb.y, qb.z, qb.w};
#pragma unroll
    for (int q = 0; q < 8; ++q) {
      const int   cc  = colBase + half * 8 + q;
      const float d2v = dv8[q];
      if (cc >= rowLo && cc < rowHi && cc != row && d2v < bd[15]) {
        float dc = d2v; int ic = cc;
#pragma unroll
        for (int s = 0; s < 16; ++s) {       // sorted bubble-insert, regs only
          const bool  sw = dc < bd[s];
          const float td = bd[s]; const int ti = bi[s];
          bd[s] = sw ? dc : td;  bi[s] = sw ? ic : ti;
          dc    = sw ? td : dc;  ic    = sw ? ti : ic;
        }
      }
    }
    asm volatile("" ::: "memory");   // keep next-iter LDS stores after reads
  }

  // merge the two sorted half-lists of each row (two-pointer, with sentinels)
#pragma unroll
  for (int s = 0; s < 16; ++s) {
    s_md[w][l15 * 34 + half * 17 + s] = bd[s];
    s_mi[w][l15 * 34 + half * 17 + s] = bi[s];
  }
  s_md[w][l15 * 34 + half * 17 + 16] = 3.38e38f;   // sentinel
  s_mi[w][l15 * 34 + half * 17 + 16] = row;
  asm volatile("s_wait_dscnt 0x0" ::: "memory");

  if (half == 0) {
    int pa = 0, pb = 0;
    for (int s = 0; s < 16; ++s) {
      const float da = s_md[w][l15 * 34 + pa];
      const float db = s_md[w][l15 * 34 + 17 + pb];
      const bool  ta = da <= db;
      d2Out[(size_t)row * KNBR + s]  = ta ? da : db;
      idxOut[(size_t)row * KNBR + s] = ta ? s_mi[w][l15 * 34 + pa]
                                          : s_mi[w][l15 * 34 + 17 + pb];
      pa += ta ? 1 : 0;  pb += ta ? 0 : 1;
    }
  }
}

// ---------------------------------------------------------------------------
// One MP layer. Per wave, per node: feat[16k x 33] @ W1 via 6x
// v_wmma_f32_16x16x32_f16 (W1 held in registers in B layout across the node
// loop), leaky, column-sum over k (shfl), 33->16 projection, GroupNorm(2x8)
// + leaky + residual — all in-wave, no LDS.
// ---------------------------------------------------------------------------
__global__ __launch_bounds__(256)
void layer_kernel(const float* __restrict__ outCur, const int* __restrict__ idx,
                  const float* __restrict__ d2, const float* __restrict__ W1,
                  const float* __restrict__ b1, const float* __restrict__ W2,
                  const float* __restrict__ b2, const float* __restrict__ gam,
                  const float* __restrict__ bet, float* __restrict__ outNext) {
  const int lane = threadIdx.x & 31;
  const int half = lane >> 4;
  const int l15  = lane & 15;
  const int gw   = blockIdx.x * (blockDim.x >> 5) + (threadIdx.x >> 5);
  const int nw   = gridDim.x * (blockDim.x >> 5);

  // W1 in WMMA-B layout (K x N tiles of 32x16): elem j<8 K=kb+8h+j, j>=8
  // K=kb+16+8h+(j-8), N = tile*16 + (lane&15). Zero-pad K>=33, N>=33.
  v16h bw[2][3];
#pragma unroll
  for (int kc = 0; kc < 2; ++kc)
#pragma unroll
    for (int t = 0; t < 3; ++t) {
      v16h bb;
#pragma unroll
      for (int j = 0; j < 16; ++j) {
        const int K  = kc * 32 + (j < 8 ? 8 * half + j : 16 + 8 * half + (j - 8));
        const int Nc = t * 16 + l15;
        const float v = (K < DIN && Nc < DIN) ? W1[K * DIN + Nc] : 0.0f;
        bb[j] = (_Float16)v;
      }
      bw[kc][t] = bb;
    }
  float cini[3];
#pragma unroll
  for (int t = 0; t < 3; ++t) {
    const int Nc = t * 16 + l15;
    cini[t] = (Nc < DIN) ? b1[Nc] : 0.0f;
  }
  float w2c[DIN];                       // this lane's output-channel column
#pragma unroll
  for (int c = 0; c < DIN; ++c) w2c[c] = W2[c * DF + l15];
  const float b2v = b2[l15], gv = gam[l15], bv = bet[l15];

  for (int i = gw; i < NPTS; i += nw) {
    const float* oi  = outCur + (size_t)i * DF + half * 8;
    const int    nbr = idx[(size_t)i * KNBR + l15];
    const float* on  = outCur + (size_t)nbr * DF + half * 8;
    const float4 oa = *(const float4*)oi,  ob = *(const float4*)(oi + 4);
    const float4 na = *(const float4*)on,  nb = *(const float4*)(on + 4);

    // A row M = k = lane&15; elems 0-7 -> feat[8h..8h+8) = out_i slice,
    // elems 8-15 -> feat[16+8h..) = out_nbr slice. Chunk1: K=32 is d2.
    v16h a0, a1;
    a0[0] = (_Float16)oa.x; a0[1] = (_Float16)oa.y;
    a0[2] = (_Float16)oa.z; a0[3] = (_Float16)oa.w;
    a0[4] = (_Float16)ob.x; a0[5] = (_Float16)ob.y;
    a0[6] = (_Float16)ob.z; a0[7] = (_Float16)ob.w;
    a0[8]  = (_Float16)na.x; a0[9]  = (_Float16)na.y;
    a0[10] = (_Float16)na.z; a0[11] = (_Float16)na.w;
    a0[12] = (_Float16)nb.x; a0[13] = (_Float16)nb.y;
    a0[14] = (_Float16)nb.z; a0[15] = (_Float16)nb.w;
#pragma unroll
    for (int e = 0; e < 16; ++e) a1[e] = (_Float16)0.0f;
    if (half == 0) a1[0] = (_Float16)d2[(size_t)i * KNBR + l15];

    float hs[3];
#pragma unroll
    for (int t = 0; t < 3; ++t) {
      v8f acc;
#pragma unroll
      for (int r = 0; r < 8; ++r) acc[r] = cini[t];
      acc = __builtin_amdgcn_wmma_f32_16x16x32_f16(
          false, a0, false, bw[0][t], (short)0, acc, false, false);
      acc = __builtin_amdgcn_wmma_f32_16x16x32_f16(
          false, a1, false, bw[1][t], (short)0, acc, false, false);
      float s = 0.f;
#pragma unroll
      for (int r = 0; r < 8; ++r) {
        float h = acc[r];
        h = h > 0.f ? h : NEGS * h;     // leaky BEFORE the k-sum
        s += h;
      }
      s += __shfl_xor(s, 16);           // fold rows 8-15 onto 0-7
      hs[t] = s;                        // col = t*16 + (lane&15)
    }

    // msg[d] = 16*b2[d] + sum_c hsum[c] * W2[c][d], d = lane&15
    float m = 16.0f * b2v;
#pragma unroll
    for (int c = 0; c < DIN; ++c) {
      const float hv = __shfl(hs[c >> 4], c & 15);
      m += hv * w2c[c];
    }

    // GroupNorm over 2 groups of 8 channels (xor-shuffle within 8 lanes)
    float mu = m;
    mu += __shfl_xor(mu, 1); mu += __shfl_xor(mu, 2); mu += __shfl_xor(mu, 4);
    mu *= 0.125f;
    float dv  = m - mu;
    float var = dv * dv;
    var += __shfl_xor(var, 1); var += __shfl_xor(var, 2); var += __shfl_xor(var, 4);
    var *= 0.125f;
    float g = dv * rsqrtf(var + 1e-5f) * gv + bv;
    g = g > 0.f ? g : NEGS * g;
    const float outv = outCur[(size_t)i * DF + l15] + g;
    if (lane < 16) outNext[(size_t)i * DF + lane] = outv;
  }
}

// ---------------------------------------------------------------------------
extern "C" void kernel_launch(void* const* d_in, const int* in_sizes, int n_in,
                              void* d_out, int out_size, void* d_ws, size_t ws_size,
                              hipStream_t stream) {
  const float* x    = (const float*)d_in[0];
  const float* y    = (const float*)d_in[1];
  const float* yat  = (const float*)d_in[2];
  const float* W1   = (const float*)d_in[3];
  const float* b1   = (const float*)d_in[4];
  const float* W2   = (const float*)d_in[5];
  const float* b2   = (const float*)d_in[6];
  const float* gam  = (const float*)d_in[7];
  const float* bet  = (const float*)d_in[8];
  const int*   xb   = (const int*)d_in[9];

  char* ws = (char*)d_ws;
  float* P      = (float*)ws;  ws += (size_t)NPTS * 8 * 4;   // 512 KB
  int*   bstart = (int*)ws;    ws += 64;
  int*   idxb   = (int*)ws;    ws += (size_t)NPTS * KNBR * 4;
  float* d2b    = (float*)ws;  ws += (size_t)NPTS * KNBR * 4;
  float* outA   = (float*)ws;  ws += (size_t)NPTS * DF * 4;
  float* outB   = (float*)ws;

  prep_P    <<<NPTS / 256, 256, 0, stream>>>(x, y, P);
  prep_batch<<<NPTS / 256, 256, 0, stream>>>(xb, bstart);
  knn_kernel<<<NPTS / 16 / KNN_WAVES, 32 * KNN_WAVES, 0, stream>>>(
      P, xb, bstart, idxb, d2b);

  float* outF = (float*)d_out;
  layer_kernel<<<512, 256, 0, stream>>>(yat,  idxb, d2b, W1,             b1,        W2,            b2,        gam,        bet,        outA);
  layer_kernel<<<512, 256, 0, stream>>>(outA, idxb, d2b, W1 + DIN*DIN,   b1 + DIN,  W2 + DIN*DF,   b2 + DF,   gam + DF,   bet + DF,   outB);
  layer_kernel<<<512, 256, 0, stream>>>(outB, idxb, d2b, W1 + 2*DIN*DIN, b1 + 2*DIN, W2 + 2*DIN*DF, b2 + 2*DF, gam + 2*DF, bet + 2*DF, outF);
}
